// KOSLM_47528108098130
// MI455X (gfx1250) — compile-verified, matrix operand order checked by reading, not run on hardware
//
#include <hip/hip_runtime.h>
#include <hip/hip_bf16.h>

// ---------------------------------------------------------------------------
// KOSLM SSM for MI455X (gfx1250): bf16 WMMA GEMMs + persistent recurrent scan
//  - parallel GEMMs (gx, Wo) use 16x64 register-blocked WMMA (A-frag reuse x4)
//  - recurrent scan: 32 persistent WGs, 4 grid barriers/step, fused z/m phase
// ---------------------------------------------------------------------------

typedef __bf16 bf16_t;
typedef __attribute__((ext_vector_type(16))) __bf16 v16bf;
typedef __attribute__((ext_vector_type(8)))  __bf16 v8bf;
typedef __attribute__((ext_vector_type(8)))  float  v8f;

#define B_  32
#define L_  512
#define D_  512
#define H_  512
#define NWG 32   // persistent workgroups for the recurrent kernel

// ---- WMMA fragment load -----------------------------------------------------
// A (16x32 bf16, row-major, ld elements) and B (32x16, stored transposed as
// Bt[n][k], same per-lane K pattern as A): lane l -> row/col = l&15,
// K-group = (l>>4)*8; elements 0..7 at k0+grp+0..7, 8..15 at k0+grp+16..23.
__device__ inline v16bf load_frag(const bf16_t* __restrict__ base, int ld,
                                  int row0, int k0) {
  int lane = threadIdx.x & 31;
  const bf16_t* p = base + (size_t)(row0 + (lane & 15)) * ld
                         + (k0 + ((lane >> 4) << 3));
  v8bf lo = *(const v8bf*)(p);
  v8bf hi = *(const v8bf*)(p + 16);
  return __builtin_shufflevector(lo, hi, 0,1,2,3,4,5,6,7,8,9,10,11,12,13,14,15);
}

__device__ inline v8f wmma_bf16(v16bf a, v16bf b, v8f c) {
  // (neg_a, A, neg_b, B, c_mod, C, reuse_a, reuse_b)
  return __builtin_amdgcn_wmma_f32_16x16x32_bf16(false, a, false, b,
                                                 (short)0, c, false, false);
}

// ---- software grid barrier (persistent kernel, agent scope) ----------------
__device__ inline void grid_sync(unsigned* cnt, unsigned* gen, unsigned nwg) {
  __syncthreads();
  if (threadIdx.x == 0) {
    __threadfence();
    unsigned g = __hip_atomic_load(gen, __ATOMIC_ACQUIRE, __HIP_MEMORY_SCOPE_AGENT);
    unsigned a = __hip_atomic_fetch_add(cnt, 1u, __ATOMIC_ACQ_REL, __HIP_MEMORY_SCOPE_AGENT);
    if (a == nwg - 1u) {
      __hip_atomic_store(cnt, 0u, __ATOMIC_RELAXED, __HIP_MEMORY_SCOPE_AGENT);
      __hip_atomic_fetch_add(gen, 1u, __ATOMIC_ACQ_REL, __HIP_MEMORY_SCOPE_AGENT);
    } else {
      while (__hip_atomic_load(gen, __ATOMIC_ACQUIRE, __HIP_MEMORY_SCOPE_AGENT) == g)
        __builtin_amdgcn_s_sleep(2);
    }
  }
  __syncthreads();
}

// ---- setup kernels ----------------------------------------------------------
__global__ void init_barrier(unsigned* bar) {
  bar[0] = 0u; bar[1] = 0u;
}

__global__ void transpose_to_bf16(const float* __restrict__ W,  // [K][N]
                                  bf16_t* __restrict__ Wt,      // [N][K]
                                  int K, int N) {
  size_t tot = (size_t)K * N;
  for (size_t i = (size_t)blockIdx.x * blockDim.x + threadIdx.x; i < tot;
       i += (size_t)gridDim.x * blockDim.x) {
    size_t k = i / N, n = i - k * N;
    Wt[n * (size_t)K + k] = (bf16_t)W[i];
  }
}

__global__ void abase_kernel(const float* __restrict__ logl,
                             float* __restrict__ ab, int n) {
  int i = blockIdx.x * blockDim.x + threadIdx.x;
  if (i < n) ab[i] = 1.0f - expf(logl[i]);
}

__global__ void rmsnorm_bf16(const float* __restrict__ x,
                             const float* __restrict__ w,
                             bf16_t* __restrict__ xn) {
  __shared__ float red[256];
  int row = blockIdx.x;
  const float* xr = x + (size_t)row * D_;
  float s = 0.f;
  for (int i = threadIdx.x; i < D_; i += 256) { float v = xr[i]; s += v * v; }
  red[threadIdx.x] = s; __syncthreads();
  for (int o = 128; o > 0; o >>= 1) {
    if (threadIdx.x < o) red[threadIdx.x] += red[threadIdx.x + o];
    __syncthreads();
  }
  float inv = rsqrtf(red[0] * (1.0f / D_) + 1e-5f);
  for (int i = threadIdx.x; i < D_; i += 256)
    xn[(size_t)row * D_ + i] = (bf16_t)(xr[i] * inv * w[i]);
}

__device__ inline void store_tile(float* __restrict__ C, int N, int rb, int col,
                                  const float* __restrict__ bias, v8f acc) {
  float bv = bias ? bias[col] : 0.0f;
#pragma unroll
  for (int j = 0; j < 8; ++j)
    C[(size_t)(rb + j) * N + col] = acc[j] + bv;
}

// ---- WMMA GEMM, 16x64 per wave: C[M,N] f32 = A[M,K]bf16 @ Bt[N,K]bf16 ------
// One A-fragment is reused across 4 B-fragments per K-step (L2 traffic /2).
__global__ void gemm_bf16_bias(const bf16_t* __restrict__ A,
                               const bf16_t* __restrict__ Bt,
                               const float* __restrict__ bias,
                               float* __restrict__ C,
                               int M, int N, int K) {
  int lane = threadIdx.x & 31;
  int gw = (blockIdx.x * blockDim.x + threadIdx.x) >> 5;
  int nw = (gridDim.x * blockDim.x) >> 5;
  int tilesN = N >> 6;                 // 64-wide column blocks
  int ntiles = (M >> 4) * tilesN;
  for (int t = gw; t < ntiles; t += nw) {
    int tm = t / tilesN, tn = t - tm * tilesN;
    int r0 = tm << 4, c0 = tn << 6;
    v8f a0 = {}, a1 = {}, a2 = {}, a3 = {};
    for (int k = 0; k < K; k += 32) {
      v16bf af = load_frag(A, K, r0, k);
      a0 = wmma_bf16(af, load_frag(Bt, K, c0,      k), a0);
      a1 = wmma_bf16(af, load_frag(Bt, K, c0 + 16, k), a1);
      a2 = wmma_bf16(af, load_frag(Bt, K, c0 + 32, k), a2);
      a3 = wmma_bf16(af, load_frag(Bt, K, c0 + 48, k), a3);
    }
    int rb = r0 + ((lane >> 4) << 3);
    int cl = lane & 15;
    store_tile(C, N, rb, c0 + cl,      bias, a0);
    store_tile(C, N, rb, c0 + 16 + cl, bias, a1);
    store_tile(C, N, rb, c0 + 32 + cl, bias, a2);
    store_tile(C, N, rb, c0 + 48 + cl, bias, a3);
  }
}

// ---- persistent recurrent scan (4 grid barriers per step) ------------------
__launch_bounds__(256)
__global__ void recurrent_kernel(
    const float*  __restrict__ gx,     // [B*L, 2H]
    const bf16_t* __restrict__ Whh_t,  // [2H][H]
    const bf16_t* __restrict__ kW1_t,  // [3H][H]
    const bf16_t* __restrict__ kW2_t,  // [H][3H]
    const float*  __restrict__ b_hh,   // [2H]
    const float*  __restrict__ k_b1,   // [3H]
    const float*  __restrict__ k_b2,   // [H]
    const float*  __restrict__ Abase,  // [H]
    bf16_t* __restrict__ hs,           // [B*L, H]
    float* c, bf16_t* cbf, float* Astate, float* cnew,
    float* innovf, bf16_t* innovbf, float* Mt, bf16_t* kmid,
    unsigned* bar_cnt, unsigned* bar_gen)
{
  int lane = threadIdx.x & 31;
  int gw   = (blockIdx.x << 3) + (threadIdx.x >> 5);   // 0..255 waves
  int tid  = blockIdx.x * 256 + threadIdx.x;           // 0..8191

  // zero initial state (h = c = A = 0)
  for (int i = tid; i < B_ * H_; i += NWG * 256) {
    c[i] = 0.f; Astate[i] = 0.f; cbf[i] = (bf16_t)0.f;
  }
  grid_sync(bar_cnt, bar_gen, NWG);

  for (int t = 0; t < L_; ++t) {
    // Phase 1 (fused): per wave, z-tile (col h) and m-tile (col h+512) of
    // g = gx_t + c @ W_hh + b_hh share one A fragment; then innov elementwise.
    {
      const int tilesN = H_ >> 4;                       // 32
      const int ntiles = (B_ >> 4) * tilesN;            // 64
      for (int tt = gw; tt < ntiles; tt += 256) {
        int tm = tt >> 5, tn = tt & 31;
        int r0 = tm << 4, c0 = tn << 4;
        v8f az = {}, am = {};
        for (int k = 0; k < H_; k += 32) {
          v16bf af = load_frag(cbf, H_, r0, k);
          az = wmma_bf16(af, load_frag(Whh_t, H_, c0,      k), az);
          am = wmma_bf16(af, load_frag(Whh_t, H_, c0 + H_, k), am);
        }
        int col = c0 + (lane & 15);
        int rb  = r0 + ((lane >> 4) << 3);
        float bz = b_hh[col], bm = b_hh[col + H_];
#pragma unroll
        for (int j = 0; j < 8; ++j) {
          int b = rb + j;
          size_t gr = ((size_t)b * L_ + t) * (2 * H_);
          float z = az[j] + bz + gx[gr + col];
          float m = tanhf(am[j] + bm + gx[gr + col + H_]);
          int e = b * H_ + col;
          float inn = z - Astate[e] * c[e] - m;
          Mt[e] = m; innovf[e] = inn; innovbf[e] = (bf16_t)inn;
        }
      }
    }
    grid_sync(bar_cnt, bar_gen, NWG);

    // Phase 2: kmid = gelu(innov @ k_W1 + b1)  (M=32, N=1536, K=512)
    {
      const int N = 3 * H_, tilesN = N >> 4, ntiles = (B_ >> 4) * tilesN; // 192
      for (int tt = gw; tt < ntiles; tt += 256) {
        int tm = tt / tilesN, tn = tt - tm * tilesN;
        int r0 = tm << 4, c0 = tn << 4;
        v8f acc = {};
        for (int k = 0; k < H_; k += 32)
          acc = wmma_bf16(load_frag(innovbf, H_, r0, k),
                          load_frag(kW1_t, H_, c0, k), acc);
        int col = c0 + (lane & 15);
        int rb  = r0 + ((lane >> 4) << 3);
        float bv = k_b1[col];
#pragma unroll
        for (int j = 0; j < 8; ++j) {
          float xv = acc[j] + bv;
          float gel = 0.5f * xv * (1.0f + erff(xv * 0.70710678118f));
          kmid[(rb + j) * N + col] = (bf16_t)gel;
        }
      }
    }
    grid_sync(bar_cnt, bar_gen, NWG);

    // Phase 3: K = 0.5*tanh(kmid @ k_W2 + b2); A_new, c_new (pre-clamp)
    {
      const int N = H_, K = 3 * H_, tilesN = N >> 4,
                ntiles = (B_ >> 4) * tilesN; // 64
      for (int tt = gw; tt < ntiles; tt += 256) {
        int tm = tt / tilesN, tn = tt - tm * tilesN;
        int r0 = tm << 4, c0 = tn << 4;
        v8f acc = {};
        for (int k = 0; k < K; k += 32)
          acc = wmma_bf16(load_frag(kmid, K, r0, k),
                          load_frag(kW2_t, K, c0, k), acc);
        int col = c0 + (lane & 15);
        int rb  = r0 + ((lane >> 4) << 3);
        float bv = k_b2[col];
        float ab = Abase[col];
#pragma unroll
        for (int j = 0; j < 8; ++j) {
          int e = (rb + j) * H_ + col;
          float Kv = tanhf(acc[j] + bv) * 0.5f;      // K_SCALE*0.5
          float An = ab * (1.0f - Kv * Kv);
          float cn = An * c[e] + Kv * innovf[e] + Mt[e];
          Astate[e] = An;
          cnew[e]   = cn;
        }
      }
    }
    grid_sync(bar_cnt, bar_gen, NWG);

    // Phase 3b: deterministic per-row norm clamp + commit state + emit h
    if (gw < B_) {
      int b = gw;
      float vals[16];
      float s = 0.f;
#pragma unroll
      for (int i = 0; i < 16; ++i) {
        float v = cnew[b * H_ + lane + i * 32];
        vals[i] = v; s += v * v;
      }
#pragma unroll
      for (int o = 16; o > 0; o >>= 1) s += __shfl_xor(s, o, 32);
      float scale = fminf(1.0f, 10.0f * rsqrtf(s + 1e-6f)); // MAX_NORM=10
#pragma unroll
      for (int i = 0; i < 16; ++i) {
        int h = lane + i * 32;
        float v = vals[i] * scale;
        c[b * H_ + h] = v;
        bf16_t vb = (bf16_t)v;
        cbf[b * H_ + h] = vb;
        hs[((size_t)b * L_ + t) * H_ + h] = vb;
      }
    }
    grid_sync(bar_cnt, bar_gen, NWG);
  }
}

// ---------------------------------------------------------------------------
extern "C" void kernel_launch(void* const* d_in, const int* in_sizes, int n_in,
                              void* d_out, int out_size, void* d_ws, size_t ws_size,
                              hipStream_t stream) {
  (void)in_sizes; (void)n_in; (void)out_size; (void)ws_size;
  const int NL = 2;
  const float* x      = (const float*)d_in[0];
  const float* norm_w = (const float*)d_in[1];
  const float* W_ih   = (const float*)d_in[2];   // [2, D, 2H]
  const float* b_ih   = (const float*)d_in[3];   // [2, 2H]
  const float* W_hh   = (const float*)d_in[4];   // [2, H, 2H]
  const float* b_hh   = (const float*)d_in[5];   // [2, 2H]
  const float* logl   = (const float*)d_in[6];   // [2, H]
  const float* k_W1   = (const float*)d_in[7];   // [2, H, 3H]
  const float* k_b1   = (const float*)d_in[8];   // [2, 3H]
  const float* k_W2   = (const float*)d_in[9];   // [2, 3H, H]
  const float* k_b2   = (const float*)d_in[10];  // [2, H]
  const float* Wo     = (const float*)d_in[11];  // [2, H, D]
  float* out = (float*)d_out;

  // ---- workspace carve-up (~113 MB) ----
  char* ws = (char*)d_ws;
  size_t off = 0;
  auto alloc = [&](size_t bytes) -> char* {
    char* p = ws + off;
    off = (off + bytes + 255) & ~(size_t)255;
    return p;
  };
  unsigned* bar    = (unsigned*)alloc(256);
  float*    Abase  = (float*)   alloc((size_t)NL * H_ * 4);
  bf16_t*   Wih_t  = (bf16_t*)  alloc((size_t)NL * 2*H_ * D_ * 2);
  bf16_t*   Whh_t  = (bf16_t*)  alloc((size_t)NL * 2*H_ * H_ * 2);
  bf16_t*   kW1_t  = (bf16_t*)  alloc((size_t)NL * 3*H_ * H_ * 2);
  bf16_t*   kW2_t  = (bf16_t*)  alloc((size_t)NL * H_ * 3*H_ * 2);
  bf16_t*   Wo_t   = (bf16_t*)  alloc((size_t)NL * D_ * H_ * 2);
  bf16_t*   xn_bf  = (bf16_t*)  alloc((size_t)B_*L_ * D_ * 2);
  float*    gx     = (float*)   alloc((size_t)B_*L_ * 2*H_ * 4);
  bf16_t*   hs_bf  = (bf16_t*)  alloc((size_t)B_*L_ * H_ * 2);
  float*    c_st   = (float*)   alloc((size_t)B_*H_ * 4);
  bf16_t*   c_bf   = (bf16_t*)  alloc((size_t)B_*H_ * 2);
  float*    A_st   = (float*)   alloc((size_t)B_*H_ * 4);
  float*    cnew   = (float*)   alloc((size_t)B_*H_ * 4);
  float*    innovf = (float*)   alloc((size_t)B_*H_ * 4);
  bf16_t*   innovb = (bf16_t*)  alloc((size_t)B_*H_ * 2);
  float*    Mt_buf = (float*)   alloc((size_t)B_*H_ * 4);
  bf16_t*   kmid   = (bf16_t*)  alloc((size_t)B_*3*H_ * 2);

  init_barrier<<<1, 1, 0, stream>>>(bar);
  abase_kernel<<<(NL*H_ + 255)/256, 256, 0, stream>>>(logl, Abase, NL*H_);

  // weight conversion to transposed bf16 (once per launch)
  for (int l = 0; l < NL; ++l) {
    transpose_to_bf16<<<1024, 256, 0, stream>>>(W_ih + (size_t)l*D_*2*H_,
        Wih_t + (size_t)l*2*H_*D_, D_, 2*H_);
    transpose_to_bf16<<<1024, 256, 0, stream>>>(W_hh + (size_t)l*H_*2*H_,
        Whh_t + (size_t)l*2*H_*H_, H_, 2*H_);
    transpose_to_bf16<<<1024, 256, 0, stream>>>(k_W1 + (size_t)l*H_*3*H_,
        kW1_t + (size_t)l*3*H_*H_, H_, 3*H_);
    transpose_to_bf16<<<1024, 256, 0, stream>>>(k_W2 + (size_t)l*3*H_*H_,
        kW2_t + (size_t)l*H_*3*H_, 3*H_, H_);
    transpose_to_bf16<<<1024, 256, 0, stream>>>(Wo + (size_t)l*H_*D_,
        Wo_t + (size_t)l*D_*H_, H_, D_);
  }

  const int M = B_ * L_; // 16384 rows
  for (int l = 0; l < NL; ++l) {
    const float* xin = (l == 0) ? x : out;
    rmsnorm_bf16<<<M, 256, 0, stream>>>(xin, norm_w, xn_bf);
    // gx = xn @ W_ih + b_ih   (16384 x 1024 x 512): 16384 16x64 wave-tiles
    gemm_bf16_bias<<<2048, 256, 0, stream>>>(xn_bf, Wih_t + (size_t)l*2*H_*D_,
        b_ih + (size_t)l*2*H_, gx, M, 2*H_, D_);
    // recurrent scan
    recurrent_kernel<<<NWG, 256, 0, stream>>>(gx,
        Whh_t + (size_t)l*2*H_*H_, kW1_t + (size_t)l*3*H_*H_,
        kW2_t + (size_t)l*H_*3*H_,
        b_hh + (size_t)l*2*H_, k_b1 + (size_t)l*3*H_, k_b2 + (size_t)l*H_,
        Abase + (size_t)l*H_, hs_bf,
        c_st, c_bf, A_st, cnew, innovf, innovb, Mt_buf, kmid,
        bar, bar + 1);
    // out = hs @ Wo   (16384 x 512 x 512): 8192 16x64 wave-tiles
    gemm_bf16_bias<<<1024, 256, 0, stream>>>(hs_bf, Wo_t + (size_t)l*D_*H_,
        (const float*)nullptr, out, M, D_, H_);
  }
}
